// RPN_41386304864520
// MI455X (gfx1250) — matrix operand order repaired and checked by reading, not compiled, not run on hardware
//
#include <hip/hip_runtime.h>
#include <stdint.h>

// ---------------------------------------------------------------------------
// CDNA5 (gfx1250) RPN implementation: WMMA implicit-GEMM conv + fused heads.
// ---------------------------------------------------------------------------

typedef __attribute__((ext_vector_type(16))) _Float16 v16h;
typedef __attribute__((ext_vector_type(8)))  float    v8f;
typedef __attribute__((ext_vector_type(4)))  unsigned u32x4;
typedef __attribute__((ext_vector_type(4)))  _Float16 v4h;
typedef int v4i __attribute__((vector_size(16)));      // matches async builtin

#define IMG_HF 800.0f
#define IMG_WF 1216.0f
#define NMS_THRESH 0.7f
#define MIN_SZ 1e-3f
#define BBOX_CLIP 4.135166556742356f   // log(1000/16)
#define NTOT 242991                    // sum over levels of HW*3

// Async global->LDS path (CDNA5 GLOBAL_LOAD_ASYNC_TO_LDS_B128, ASYNCcnt).
#if defined(__has_builtin)
#if __has_builtin(__builtin_amdgcn_global_load_async_to_lds_b128) && \
    __has_builtin(__builtin_amdgcn_s_wait_asynccnt)
#define USE_ASYNC_LDS 1
#endif
#endif
#ifndef USE_ASYNC_LDS
#define USE_ASYNC_LDS 0
#endif

#if USE_ASYNC_LDS
typedef __attribute__((address_space(1))) v4i gv4i;    // global (__device__)
typedef __attribute__((address_space(3))) v4i lv4i;    // LDS (__shared__)
#endif

__constant__ int   c_W[5]    = {304,152,76,38,19};
__constant__ int   c_N[5]    = {182400,45600,11400,2850,741};
__constant__ int   c_off[5]  = {0,182400,228000,239400,242250};
__constant__ int   c_K[5]    = {1000,1000,1000,1000,741};
__constant__ float c_size[5] = {32.f,64.f,128.f,256.f,512.f};
__constant__ int   c_sh[5]   = {4,8,16,32,61};   // 800  // H
__constant__ int   c_sw[5]   = {4,8,16,32,64};   // 1216 // W

__device__ __forceinline__ unsigned enc_f(float f) {
  unsigned u = __float_as_uint(f);
  return (u & 0x80000000u) ? ~u : (u | 0x80000000u);
}
__device__ __forceinline__ float dec_f(unsigned k) {
  unsigned u = (k & 0x80000000u) ? (k ^ 0x80000000u) : ~k;
  return __uint_as_float(u);
}

// ---------------------------------------------------------------------------
// K1: repack conv weights -> f16 chunk-major [ci=72][n=256][kk=32]
//     (k = r*256 + c, r = ky*3+kx, ci = r*8+cc, c = cc*32+kk)
//     and fused head weights -> f16 [n=16][k=256] (n: 0-2 cls, 3-14 box, 15 pad)
// ---------------------------------------------------------------------------
__global__ void prep_weights(const float* __restrict__ wconv,
                             const float* __restrict__ wcls,
                             const float* __restrict__ wbox,
                             _Float16* __restrict__ wT2,
                             _Float16* __restrict__ wHead2) {
  int t = blockIdx.x * blockDim.x + threadIdx.x;
  if (t < 72 * 256 * 32) {
    int kk = t & 31;
    int n  = (t >> 5) & 255;
    int ci = t >> 13;
    int r  = ci >> 3, cc = ci & 7;
    int c  = cc * 32 + kk;
    int ky = r / 3, kx = r % 3;
    wT2[t] = (_Float16)wconv[(((size_t)n * 256 + c) * 3 + ky) * 3 + kx];
  } else {
    int u = t - 72 * 256 * 32;
    if (u < 16 * 256) {
      int k = u & 255;
      int n = u >> 8;
      float v = 0.f;
      if (n < 3)       v = wcls[n * 256 + k];
      else if (n < 15) v = wbox[(n - 3) * 256 + k];
      wHead2[u] = (_Float16)v;
    }
  }
}

// ---------------------------------------------------------------------------
// K2: implicit-GEMM 3x3 conv (WMMA f16 16x16x32, K=2304) + ReLU, then fused
//     1x1 cls/box heads via a second WMMA GEMM (K=256) on waves 0-3.
//     Block = 512 threads = 16 wave32, one 64-pixel tile per block.
//     Each wave: 4 accumulators (4 m-subtiles x its n-tile) -> 4 WMMAs/chunk.
// ---------------------------------------------------------------------------
__global__ __launch_bounds__(512) void rpn_head(
    const float* __restrict__ feat, const _Float16* __restrict__ wT2,
    const _Float16* __restrict__ wHead2,
    const float* __restrict__ bconv, const float* __restrict__ bcls,
    const float* __restrict__ bbox,
    float* __restrict__ scores, float* __restrict__ deltas,
    int H, int W, int HW, int lvl_off, int tiles) {
  __shared__ alignas(16) _Float16 a_lds[64 * 32];    // [m][kk]
  __shared__ alignas(16) _Float16 b_lds[256 * 32];   // [n][kk]
  __shared__ alignas(16) _Float16 t_lds[64 * 272];   // [m][n], padded stride

  const int b    = blockIdx.x / tiles;
  const int p0   = (blockIdx.x % tiles) * 64;
  const int tid  = threadIdx.x;
  const int wave = tid >> 5;
  const int lane = tid & 31;
  const int hhi  = lane >> 4;   // 0/1 (lane group)
  const int l16  = lane & 15;

  // A-staging thread mapping: thread -> (pixel am, 4 contiguous channels
  // starting at khi*4). Consecutive tid -> consecutive pixel -> coalesced x
  // reads per tap; the 4 halves pack into a single ds_store_b64.
  const int am  = tid & 63;     // pixel in tile
  const int khi = tid >> 6;     // 0..7 -> channel base khi*4 within chunk
  const int ap  = p0 + am;
  const int ay  = ap / W, ax = ap % W;            // hoisted out of all loops
  const size_t HWs = (size_t)H * W;
  const float* featb = feat + (size_t)b * 256 * HWs + (size_t)(khi * 4) * HWs;

  v8f acc0 = {}, acc1 = {}, acc2 = {}, acc3 = {};

  union Frag { v16h v; u32x4 u[2]; };

  for (int r = 0; r < 9; ++r) {                   // 3x3 tap (geometry per tap)
    const int dy = r / 3 - 1, dx = r % 3 - 1;
    const int yy = ay + dy, xx = ax + dx;
    const bool ok = (ap < HW) && (yy >= 0) && (yy < H) && (xx >= 0) && (xx < W);
    // Halo/tail handling WITHOUT divergent loads: clamp address into range
    // (always valid) and zero via a multiplicative mask after the load. This
    // lets the 4 strided channel loads issue as one clause w/ a single wait.
    const int yyc = min(max(yy, 0), H - 1);
    const int xxc = min(max(xx, 0), W - 1);
    const float msk = ok ? 1.0f : 0.0f;
    const float* fb = featb + (size_t)yyc * W + xxc;
    for (int cc = 0; cc < 8; ++cc) {              // channel chunk of 32
      const int ci = r * 8 + cc;
      // --- stage A: 64x32 f16 patch (fp32 -> f16, masked halo) ---
      {
        float f0 = fb[(size_t)(cc * 32 + 0) * HWs];
        float f1 = fb[(size_t)(cc * 32 + 1) * HWs];
        float f2 = fb[(size_t)(cc * 32 + 2) * HWs];
        float f3 = fb[(size_t)(cc * 32 + 3) * HWs];
        v4h pk;
        pk[0] = (_Float16)(f0 * msk);
        pk[1] = (_Float16)(f1 * msk);
        pk[2] = (_Float16)(f2 * msk);
        pk[3] = (_Float16)(f3 * msk);
        *(v4h*)(a_lds + am * 32 + khi * 4) = pk;   // one ds_store_b64
      }
      // --- stage B: 16KB chunk -> LDS ---
#if USE_ASYNC_LDS
      {
        // CDNA5 async copy: global -> LDS without VGPR round-trip (ASYNCcnt)
        lv4i* dst = (lv4i*)(b_lds + tid * 16);
        gv4i* src = (gv4i*)(wT2 + (size_t)ci * 8192 + tid * 16);
        __builtin_amdgcn_global_load_async_to_lds_b128(src, dst, 0, 0);
        __builtin_amdgcn_global_load_async_to_lds_b128(src + 1, dst + 1, 0, 0);
        if (ci + 1 < 72) __builtin_prefetch(wT2 + (size_t)(ci + 1) * 8192, 0, 1);
        __builtin_amdgcn_s_wait_asynccnt(0);
      }
#else
      {
        const u32x4* src = (const u32x4*)(wT2 + (size_t)ci * 8192);
        u32x4* dst = (u32x4*)b_lds;
        dst[tid * 2 + 0] = src[tid * 2 + 0];
        dst[tid * 2 + 1] = src[tid * 2 + 1];
        if (ci + 1 < 72) __builtin_prefetch(wT2 + (size_t)(ci + 1) * 8192, 0, 1);
      }
#endif
      __syncthreads();
      // --- build fragments (ds_load_b128) and 4 independent WMMAs ---
      Frag Bf;
      {
        const u32x4* bq =
            (const u32x4*)(b_lds + (wave * 16 + l16) * 32 + hhi * 8);
        Bf.u[0] = bq[0];
        Bf.u[1] = bq[2];
      }
      Frag A0, A1, A2, A3;
      {
        const u32x4* aq0 = (const u32x4*)(a_lds + (0 * 16 + l16) * 32 + hhi * 8);
        const u32x4* aq1 = (const u32x4*)(a_lds + (1 * 16 + l16) * 32 + hhi * 8);
        const u32x4* aq2 = (const u32x4*)(a_lds + (2 * 16 + l16) * 32 + hhi * 8);
        const u32x4* aq3 = (const u32x4*)(a_lds + (3 * 16 + l16) * 32 + hhi * 8);
        A0.u[0] = aq0[0]; A0.u[1] = aq0[2];
        A1.u[0] = aq1[0]; A1.u[1] = aq1[2];
        A2.u[0] = aq2[0]; A2.u[1] = aq2[2];
        A3.u[0] = aq3[0]; A3.u[1] = aq3[2];
      }
      acc0 = __builtin_amdgcn_wmma_f32_16x16x32_f16(false, A0.v, false, Bf.v,
                                                    (short)0, acc0, false, false);
      acc1 = __builtin_amdgcn_wmma_f32_16x16x32_f16(false, A1.v, false, Bf.v,
                                                    (short)0, acc1, false, false);
      acc2 = __builtin_amdgcn_wmma_f32_16x16x32_f16(false, A2.v, false, Bf.v,
                                                    (short)0, acc2, false, false);
      acc3 = __builtin_amdgcn_wmma_f32_16x16x32_f16(false, A3.v, false, Bf.v,
                                                    (short)0, acc3, false, false);
      __syncthreads();
    }
  }

  // bias + ReLU -> t_lds (f16, padded stride 272 to dodge bank conflicts)
  {
    const int ncol = wave * 16 + l16;
    const float bias = bconv[ncol];
    v8f accs[4] = {acc0, acc1, acc2, acc3};
#pragma unroll
    for (int mt = 0; mt < 4; ++mt) {
#pragma unroll
      for (int v = 0; v < 8; ++v) {
        int m = mt * 16 + v + hhi * 8;
        float t = accs[mt][v] + bias;
        t_lds[m * 272 + ncol] = (_Float16)(t > 0.f ? t : 0.f);
      }
    }
  }
  __syncthreads();

  // Fused heads: waves 0-3 compute [16 x 16] = t[16x256] * wHead[256x16]
  if (wave < 4) {
    v8f acch = {};
#pragma unroll
    for (int kc = 0; kc < 8; ++kc) {
      Frag Af, Bf;
      const u32x4* aq =
          (const u32x4*)(t_lds + (wave * 16 + l16) * 272 + kc * 32 + hhi * 8);
      Af.u[0] = aq[0];
      Af.u[1] = aq[2];
      const u32x4* bq = (const u32x4*)(wHead2 + l16 * 256 + kc * 32 + hhi * 8);
      Bf.u[0] = bq[0];
      Bf.u[1] = bq[2];
      acch = __builtin_amdgcn_wmma_f32_16x16x32_f16(false, Af.v, false, Bf.v,
                                                    (short)0, acch, false, false);
    }
    const int n = l16;
#pragma unroll
    for (int v = 0; v < 8; ++v) {
      int m = wave * 16 + v + hhi * 8;
      int p = p0 + m;
      if (p < HW) {
        float val = acch[v];
        if (n < 3) {
          scores[(size_t)b * NTOT + lvl_off + p * 3 + n] = val + bcls[n];
        } else if (n < 15) {
          int o = n - 3;
          int a = o >> 2, j = o & 3;
          deltas[((size_t)b * NTOT + lvl_off + p * 3 + a) * 4 + j] = val + bbox[o];
        }
      }
    }
  }
}

// ---------------------------------------------------------------------------
// K3: per (image,level) top-K (sorted desc, tie idx asc):
//     radix-select on flip-float keys + compaction + 1024-wide bitonic sort.
// ---------------------------------------------------------------------------
__global__ __launch_bounds__(1024) void topk_kernel(const float* __restrict__ scores,
                                                    float* __restrict__ topv,
                                                    int* __restrict__ topi) {
  const int g   = blockIdx.x;
  const int lvl = g % 5;
  const int b   = g / 5;
  const int N   = c_N[lvl];
  const int K   = c_K[lvl];
  const float* s = scores + (size_t)b * NTOT + c_off[lvl];

  __shared__ unsigned hist[256];
  __shared__ unsigned long long pairs[1024];
  __shared__ unsigned s_sel, s_cnt, cG, cE;
  const int tid = threadIdx.x;

  unsigned prefix = 0, prefmask = 0;
  unsigned kneed = (unsigned)K;
  for (int pass = 0; pass < 4; ++pass) {
    const int shift = 24 - pass * 8;
    if (tid < 256) hist[tid] = 0;
    __syncthreads();
    for (int i = tid; i < N; i += 1024) {
      unsigned key = enc_f(s[i]);
      if ((key & prefmask) == prefix)
        atomicAdd(&hist[(key >> shift) & 255], 1u);
    }
    __syncthreads();
    if (tid == 0) {
      unsigned cum = 0, sel = 0;
      for (int bin = 255; bin >= 0; --bin) {
        unsigned h = hist[bin];
        if (cum + h >= kneed) { sel = (unsigned)bin; break; }
        cum += h;
      }
      s_sel = sel; s_cnt = cum;
    }
    __syncthreads();
    prefix  |= (s_sel << shift);
    prefmask |= (0xFFu << shift);
    kneed   -= s_cnt;
    __syncthreads();
  }
  const unsigned T = prefix;
  const unsigned countGreater = (unsigned)K - kneed;
  if (tid == 0) { cG = 0; cE = 0; }
  pairs[tid] = 0ull;
  __syncthreads();
  for (int i = tid; i < N; i += 1024) {
    unsigned key = enc_f(s[i]);
    if (key > T) {
      unsigned pos = atomicAdd(&cG, 1u);
      pairs[pos] = ((unsigned long long)key << 32) | (unsigned)(~(unsigned)i);
    } else if (key == T) {
      unsigned e = atomicAdd(&cE, 1u);
      if (e < kneed)
        pairs[countGreater + e] =
            ((unsigned long long)key << 32) | (unsigned)(~(unsigned)i);
    }
  }
  __syncthreads();
  // bitonic sort, descending (key desc, idx asc via ~idx in low bits)
  for (unsigned size = 2; size <= 1024; size <<= 1) {
    for (unsigned stride = size >> 1; stride > 0; stride >>= 1) {
      unsigned i = (unsigned)tid, j = i ^ stride;
      if (j > i) {
        unsigned long long a = pairs[i], bb = pairs[j];
        bool desc = ((i & size) == 0);
        if ((a < bb) == desc) { pairs[i] = bb; pairs[j] = a; }
      }
      __syncthreads();
    }
  }
  if (tid < 1000) {
    if (tid < K) {
      unsigned long long pr = pairs[tid];
      topv[g * 1000 + tid] = dec_f((unsigned)(pr >> 32));
      topi[g * 1000 + tid] = (int)(~(unsigned)pr);
    } else {
      topv[g * 1000 + tid] = -1e30f;
      topi[g * 1000 + tid] = 0;
    }
  }
}

// ---------------------------------------------------------------------------
// K4: gather deltas, build anchors analytically, decode, clip, sigmoid.
// ---------------------------------------------------------------------------
__global__ __launch_bounds__(1024) void decode_kernel(
    const float* __restrict__ topv, const int* __restrict__ topi,
    const float* __restrict__ deltas, float* __restrict__ boxes,
    float* __restrict__ probs) {
  const int g   = blockIdx.x;
  const int lvl = g % 5;
  const int b   = g / 5;
  const int K   = c_K[lvl];
  const int i   = threadIdx.x;
  if (i >= 1000) return;
  if (i >= K) {
    probs[g * 1000 + i] = -1.0f;
    ((float4*)boxes)[g * 1000 + i] = make_float4(0.f, 0.f, 0.f, 0.f);
    return;
  }
  const int idx = topi[g * 1000 + i];
  const float v = topv[g * 1000 + i];
  const int a = idx % 3;
  const int p = idx / 3;
  const int W = c_W[lvl];
  const int y = p / W, x = p % W;
  // torchvision base anchors: round(±w/2, ±h/2) (rint = round-half-even)
  const float ratio = (a == 0) ? 0.5f : (a == 1 ? 1.0f : 2.0f);
  const float hr = sqrtf(ratio), wr = 1.0f / hr;
  const float sz = c_size[lvl];
  const float wsz = wr * sz, hsz = hr * sz;
  const float bx1 = rintf(-wsz * 0.5f), by1 = rintf(-hsz * 0.5f);
  const float bx2 = rintf(wsz * 0.5f),  by2 = rintf(hsz * 0.5f);
  const float sx = (float)(x * c_sw[lvl]);
  const float sy = (float)(y * c_sh[lvl]);
  const float ax1 = sx + bx1, ay1 = sy + by1, ax2 = sx + bx2, ay2 = sy + by2;
  const float wa = ax2 - ax1, ha = ay2 - ay1;
  const float cxa = ax1 + 0.5f * wa, cya = ay1 + 0.5f * ha;
  const size_t df = ((size_t)b * NTOT + c_off[lvl] + idx) * 4;
  float dx = deltas[df + 0], dyv = deltas[df + 1];
  float dw = fminf(deltas[df + 2], BBOX_CLIP);
  float dh = fminf(deltas[df + 3], BBOX_CLIP);
  const float cx = dx * wa + cxa, cy = dyv * ha + cya;
  const float w = expf(dw) * wa, h = expf(dh) * ha;
  const float x1 = fminf(fmaxf(cx - 0.5f * w, 0.f), IMG_WF);
  const float y1 = fminf(fmaxf(cy - 0.5f * h, 0.f), IMG_HF);
  const float x2 = fminf(fmaxf(cx + 0.5f * w, 0.f), IMG_WF);
  const float y2 = fminf(fmaxf(cy + 0.5f * h, 0.f), IMG_HF);
  ((float4*)boxes)[g * 1000 + i] = make_float4(x1, y1, x2, y2);
  probs[g * 1000 + i] = 1.0f / (1.0f + expf(-v));
}

// ---------------------------------------------------------------------------
// K5: greedy sequential NMS over 1000 score-sorted boxes (per image,level).
// ---------------------------------------------------------------------------
__global__ __launch_bounds__(1024) void nms_kernel(const float* __restrict__ boxes,
                                                   float* __restrict__ probs) {
  const int g   = blockIdx.x;
  const int K   = c_K[g % 5];
  __shared__ float4 s_box[1000];
  __shared__ int s_flag;
  const int tid = threadIdx.x;
  float4 mb = make_float4(0.f, 0.f, 0.f, 0.f);
  float myprob = -1.f;
  bool valid = false;
  if (tid < K) {
    mb = ((const float4*)boxes)[g * 1000 + tid];
    myprob = probs[g * 1000 + tid];
    valid = ((mb.z - mb.x) >= MIN_SZ) && ((mb.w - mb.y) >= MIN_SZ);
    s_box[tid] = mb;
  }
  __syncthreads();
  const float myarea = (mb.z - mb.x) * (mb.w - mb.y);
  bool sup = false, mykeep = false;
  for (int i = 0; i < K; ++i) {
    if (tid == i) { mykeep = valid && !sup; s_flag = mykeep ? 1 : 0; }
    __syncthreads();
    if (s_flag && tid > i && tid < K) {
      float4 ob = s_box[i];
      float lx = fmaxf(mb.x, ob.x), ly = fmaxf(mb.y, ob.y);
      float rx = fminf(mb.z, ob.z), ry = fminf(mb.w, ob.w);
      float iw = fmaxf(rx - lx, 0.f), ih = fmaxf(ry - ly, 0.f);
      float inter = iw * ih;
      float oarea = (ob.z - ob.x) * (ob.w - ob.y);
      float iou = inter / (myarea + oarea - inter + 1e-9f);
      if (iou > NMS_THRESH) sup = true;
    }
    __syncthreads();
  }
  if (tid < K) probs[g * 1000 + tid] = mykeep ? myprob : -1.0f;
}

// ---------------------------------------------------------------------------
// K6: global top-1000 over 5000-entry concat by exact stable rank; zero boxes
//     with score < 0. One block per image.
// ---------------------------------------------------------------------------
__global__ __launch_bounds__(1024) void final_topk(const float* __restrict__ probs,
                                                   const float* __restrict__ boxes,
                                                   float* __restrict__ out) {
  const int b = blockIdx.x;
  __shared__ float s_sc[5000];
  const int tid = threadIdx.x;
  for (int i = tid; i < 5000; i += 1024) s_sc[i] = probs[b * 5000 + i];
  __syncthreads();
  for (int i = tid; i < 5000; i += 1024) {
    const float si = s_sc[i];
    int r = 0;
    for (int j = 0; j < 5000; ++j) {
      float sj = s_sc[j];
      r += (sj > si) || (sj == si && j < i);
    }
    if (r < 1000) {
      float4 bx = make_float4(0.f, 0.f, 0.f, 0.f);
      if (si >= 0.f) bx = ((const float4*)boxes)[b * 5000 + i];
      ((float4*)out)[b * 1000 + r] = bx;
    }
  }
}

// ---------------------------------------------------------------------------
extern "C" void kernel_launch(void* const* d_in, const int* in_sizes, int n_in,
                              void* d_out, int out_size, void* d_ws, size_t ws_size,
                              hipStream_t stream) {
  (void)in_sizes; (void)n_in; (void)out_size; (void)ws_size;
  const float* feat[5] = {(const float*)d_in[0], (const float*)d_in[1],
                          (const float*)d_in[2], (const float*)d_in[3],
                          (const float*)d_in[4]};
  const float* wconv = (const float*)d_in[5];
  const float* bconv = (const float*)d_in[6];
  const float* wcls  = (const float*)d_in[7];
  const float* bcls  = (const float*)d_in[8];
  const float* wbox  = (const float*)d_in[9];
  const float* bbox  = (const float*)d_in[10];

  char* ws = (char*)d_ws;
  size_t off = 0;
  auto alloc = [&](size_t bytes) -> void* {
    off = (off + 255) & ~(size_t)255;
    void* p = ws + off;
    off += bytes;
    return p;
  };
  _Float16* wT2    = (_Float16*)alloc((size_t)72 * 256 * 32 * 2);
  _Float16* wHead2 = (_Float16*)alloc((size_t)16 * 256 * 2);
  float* scores    = (float*)alloc((size_t)2 * NTOT * 4);
  float* deltas    = (float*)alloc((size_t)2 * NTOT * 4 * 4);
  float* topv      = (float*)alloc((size_t)10 * 1000 * 4);
  int*   topi      = (int*)alloc((size_t)10 * 1000 * 4);
  float* boxes     = (float*)alloc((size_t)10 * 1000 * 4 * 4);
  float* probs     = (float*)alloc((size_t)10 * 1000 * 4);

  prep_weights<<<(72 * 256 * 32 + 16 * 256) / 256, 256, 0, stream>>>(
      wconv, wcls, wbox, wT2, wHead2);

  const int H[5]  = {200, 100, 50, 25, 13};
  const int W[5]  = {304, 152, 76, 38, 19};
  const int HW[5] = {60800, 15200, 3800, 950, 247};
  const int OF[5] = {0, 182400, 228000, 239400, 242250};
  for (int l = 0; l < 5; ++l) {
    int tiles = (HW[l] + 63) / 64;
    rpn_head<<<2 * tiles, 512, 0, stream>>>(feat[l], wT2, wHead2, bconv, bcls,
                                            bbox, scores, deltas, H[l], W[l],
                                            HW[l], OF[l], tiles);
  }
  topk_kernel<<<10, 1024, 0, stream>>>(scores, topv, topi);
  decode_kernel<<<10, 1024, 0, stream>>>(topv, topi, deltas, boxes, probs);
  nms_kernel<<<10, 1024, 0, stream>>>(boxes, probs);
  final_topk<<<2, 1024, 0, stream>>>(probs, boxes, (float*)d_out);
}
